// ParallelTransformerBlock_44504451121229
// MI455X (gfx1250) — compile-verified, matrix operand order checked by reading, not run on hardware
//
#include <hip/hip_runtime.h>
#include <hip/hip_bf16.h>

typedef __bf16 bf16;
typedef __attribute__((ext_vector_type(16))) __bf16 v16bf;
typedef __attribute__((ext_vector_type(8)))  float  v8f;
typedef unsigned int u32x4 __attribute__((ext_vector_type(4)));
typedef int i32x8 __attribute__((ext_vector_type(8)));
typedef int i32x4 __attribute__((ext_vector_type(4)));

static constexpr int Bc  = 2;
static constexpr int Sc  = 2048;
static constexpr int Dc  = 2048;
static constexpr int Hc  = 16;
static constexpr int DHc = 128;
static constexpr int FFc = 8192;
static constexpr int NFc = 3 * Dc + FFc;   // 14336
static constexpr float EPSc = 1e-6f;

#if defined(__HIP_DEVICE_COMPILE__) && __has_builtin(__builtin_amdgcn_tensor_load_to_lds)
#define HAVE_TDM 1
#else
#define HAVE_TDM 0
#endif

__device__ __forceinline__ v8f zero8() {
  v8f z = {0.f, 0.f, 0.f, 0.f, 0.f, 0.f, 0.f, 0.f};
  return z;
}

// Load a 16x32 (A) or 32x16 (B) bf16 WMMA fragment. `p` points at element
// (row_of_this_lane, k_base_of_this_lane); halves are p[0..7] and p[16..23].
__device__ __forceinline__ v16bf frag_ld(const bf16* p) {
  union { v16bf v; uint4 q[2]; } u;
  u.q[0] = *(const uint4*)p;
  u.q[1] = *(const uint4*)(p + 16);
  return u.v;
}

#if HAVE_TDM
// Issue one TDM 2-D tile load: tile_k x tile_rows bf16 elements, row stride
// `row_stride_elems` in global memory, destination LDS at byte offset
// `lds_addr`, with 16B of LDS padding inserted after every 128B of data
// (pad_interval code 4 = 32 DWORDs, pad_amount code 3 = 4 DWORDs).
__device__ __forceinline__ void tdm_load_2d(unsigned lds_addr, const bf16* gptr,
                                            int tile_k, int tile_rows,
                                            int row_stride_elems) {
  unsigned long long ga = (unsigned long long)(size_t)gptr;
  u32x4 g0;
  g0.x = 1u;                                        // count=1, user descriptor
  g0.y = lds_addr;                                  // lds_addr [63:32]
  g0.z = (unsigned)ga;                              // global_addr lo
  g0.w = (unsigned)((ga >> 32) & 0x01FFFFFFull) | 0x80000000u;  // type=2 ("image")
  i32x8 g1;
  g1[0] = (1 << 16) | (1 << 20) | (4 << 22) | (3 << 25); // data_size=2B, pad en/intv/amt
  g1[1] = 0;                                        // atomic_barrier=0, tdim0 lo16=0
  g1[2] = 0x4000;                                   // tensor_dim0 = 1<<30 (hi16)
  g1[3] = 0x4000 | (tile_k << 16);                  // tensor_dim1 = 1<<30, tile_dim0
  g1[4] = tile_rows;                                // tile_dim1; tile_dim2 = 0
  g1[5] = row_stride_elems;                         // tensor_dim0_stride lo32
  g1[6] = 0;
  g1[7] = 0;
  i32x4 gz = {0, 0, 0, 0};
#if __clang_major__ >= 23
  i32x8 gz8 = {0, 0, 0, 0, 0, 0, 0, 0};
  __builtin_amdgcn_tensor_load_to_lds(g0, g1, gz, gz, gz8, 0);
#else
  __builtin_amdgcn_tensor_load_to_lds(g0, g1, gz, gz, 0);
#endif
}
#endif

// ---------------------------------------------------------------------------
// f32 -> bf16 conversion (grid-stride)
// ---------------------------------------------------------------------------
__global__ void cvt_bf16_kernel(const float* __restrict__ src,
                                bf16* __restrict__ dst, size_t n) {
  size_t i = (size_t)blockIdx.x * blockDim.x + threadIdx.x;
  size_t stride = (size_t)gridDim.x * blockDim.x;
  for (; i < n; i += stride) dst[i] = (bf16)src[i];
}

// ---------------------------------------------------------------------------
// Tiled transpose + f32->bf16:  dst[N,K] = (bf16) src[K,N]^T
// Block 256 threads handles a 32x32 tile.
// ---------------------------------------------------------------------------
__global__ __launch_bounds__(256) void transpose_cvt_kernel(
    const float* __restrict__ src, bf16* __restrict__ dst, int K, int N) {
  __shared__ float t[32][33];
  const int n0 = blockIdx.x * 32, k0 = blockIdx.y * 32;
  const int tx = threadIdx.x & 31, ty = threadIdx.x >> 5;   // ty: 0..7
#pragma unroll
  for (int i = ty; i < 32; i += 8)
    t[i][tx] = src[(size_t)(k0 + i) * N + n0 + tx];
  __syncthreads();
#pragma unroll
  for (int i = ty; i < 32; i += 8)
    dst[(size_t)(n0 + i) * K + k0 + tx] = (bf16)t[tx][i];
}

// ---------------------------------------------------------------------------
// NT GEMM, bf16 in / f32 accumulate:  C[M,N] = A[M,K] @ Bt[N,K]^T + bias
// 256 threads (8 waves), tile 128x128x64, wave tile 64x32 (4x2 frags).
// Tiles staged by the Tensor Data Mover into ping-pong LDS buffers; the DMA
// for tile i+1 overlaps WMMA work on tile i (TENSORcnt + workgroup barrier).
// ---------------------------------------------------------------------------
static constexpr int BKg = 64;   // staged k per step (2 WMMA k-chunks)
static constexpr int LDK = 72;   // padded LDS row stride in elements (144B)

__global__ __launch_bounds__(256) void gemm_bf16_nt_kernel(
    const bf16* __restrict__ A, const bf16* __restrict__ Bt,
    const float* __restrict__ bias, float* __restrict__ C,
    int M, int N, int K) {
  __shared__ bf16 As[2][128][LDK];
  __shared__ bf16 Bs[2][128][LDK];

  const int tid  = threadIdx.x;
  const int lane = tid & 31;
  const int wave = tid >> 5;
  const int bm = blockIdx.y * 128;
  const int bn = blockIdx.x * 128;
  const int wm = (wave >> 2) * 64;     // 0 or 64
  const int wn = (wave & 3) * 32;      // 0,32,64,96
  const int lrow = lane & 15;
  const int lkb  = (lane >> 4) * 8;

  v8f acc[4][2];
#pragma unroll
  for (int i = 0; i < 4; ++i)
#pragma unroll
    for (int j = 0; j < 2; ++j) acc[i][j] = zero8();

  const int nsteps = K / BKg;

#if HAVE_TDM
  if (wave == 0) {
    tdm_load_2d((unsigned)(size_t)&As[0][0][0], A  + (size_t)bm * K, BKg, 128, K);
    tdm_load_2d((unsigned)(size_t)&Bs[0][0][0], Bt + (size_t)bn * K, BKg, 128, K);
    __builtin_amdgcn_s_wait_tensorcnt(0);
  }
  __syncthreads();
#endif

  for (int step = 0; step < nsteps; ++step) {
    const int cur = step & 1;
#if HAVE_TDM
    if (wave == 0 && step + 1 < nsteps) {
      const int k1 = (step + 1) * BKg;
      tdm_load_2d((unsigned)(size_t)&As[cur ^ 1][0][0], A  + (size_t)bm * K + k1, BKg, 128, K);
      tdm_load_2d((unsigned)(size_t)&Bs[cur ^ 1][0][0], Bt + (size_t)bn * K + k1, BKg, 128, K);
    }
#else
    // Fallback: manual staging into buffer `cur`
    __syncthreads();
#pragma unroll
    for (int it = 0; it < 4; ++it) {
      int idx = tid + it * 256;
      int r = idx >> 3, seg = (idx & 7) * 8;
      *(uint4*)&As[cur][r][seg] = *(const uint4*)(A  + (size_t)(bm + r) * K + step * BKg + seg);
      *(uint4*)&Bs[cur][r][seg] = *(const uint4*)(Bt + (size_t)(bn + r) * K + step * BKg + seg);
    }
    __syncthreads();
#endif

#pragma unroll
    for (int kc = 0; kc < 2; ++kc) {
      v16bf afr[4], bfr[2];
#pragma unroll
      for (int i = 0; i < 4; ++i)
        afr[i] = frag_ld(&As[cur][wm + i * 16 + lrow][kc * 32 + lkb]);
#pragma unroll
      for (int j = 0; j < 2; ++j)
        bfr[j] = frag_ld(&Bs[cur][wn + j * 16 + lrow][kc * 32 + lkb]);
#pragma unroll
      for (int i = 0; i < 4; ++i)
#pragma unroll
        for (int j = 0; j < 2; ++j)
          acc[i][j] = __builtin_amdgcn_wmma_f32_16x16x32_bf16(
              false, afr[i], false, bfr[j], (short)0, acc[i][j], false, false);
    }

#if HAVE_TDM
    if (wave == 0) __builtin_amdgcn_s_wait_tensorcnt(0);
    __syncthreads();
#endif
  }

  // ---- epilogue: bias add, f32 store ----
  const int lhi = (lane >> 4) * 8;
#pragma unroll
  for (int i = 0; i < 4; ++i) {
    int row0 = bm + wm + i * 16 + lhi;
#pragma unroll
    for (int j = 0; j < 2; ++j) {
      int col = bn + wn + j * 16 + lrow;
      float bv = bias ? bias[col] : 0.f;
#pragma unroll
      for (int g = 0; g < 8; ++g)
        C[(size_t)(row0 + g) * N + col] = acc[i][j][g] + bv;
    }
  }
}

// ---------------------------------------------------------------------------
// Split fused -> q,k,v with RoPE (reference's rotate_half has NO sign flip),
// q scaled by DH^-0.5. Outputs bf16 in [B,H,S,DH].
// ---------------------------------------------------------------------------
__global__ __launch_bounds__(256) void qkv_rope_kernel(
    const float* __restrict__ fused, const float* __restrict__ rope_w,
    bf16* __restrict__ Qb, bf16* __restrict__ Kb, bf16* __restrict__ Vb) {
  const int row = blockIdx.x;          // b*S + s
  const int b = row / Sc, s = row % Sc;
  const float* f = fused + (size_t)row * NFc;
  const float qscale = 0.088388347648318447f;  // 128^-0.5

  for (int j = threadIdx.x; j < Dc; j += blockDim.x) {
    const int h = j >> 7, d = j & 127;
    const float pe = rope_w[s * DHc + d];
    const float c = __cosf(pe), sn = __sinf(pe);
    const int dp = (d < 64) ? d + 64 : d - 64;
    const int hb = h * DHc;
    const float q  = f[hb + d],  k  = f[Dc + hb + d];
    const float qp = f[hb + dp], kp = f[Dc + hb + dp];
    const float v  = f[2 * Dc + hb + d];
    const size_t idx = (((size_t)b * Hc + h) * Sc + s) * DHc + d;
    Qb[idx] = (bf16)((q * c + qp * sn) * qscale);
    Kb[idx] = (bf16)(k * c + kp * sn);
    Vb[idx] = (bf16)v;
  }
}

// ---------------------------------------------------------------------------
// LayerNorm(gamma1,beta1) + ReLU over the FF slice of fused -> bf16 h
// ---------------------------------------------------------------------------
__global__ __launch_bounds__(256) void ln_relu_kernel(
    const float* __restrict__ fused, const float* __restrict__ gamma,
    const float* __restrict__ beta, bf16* __restrict__ Hb) {
  const int row = blockIdx.x;
  const float* f = fused + (size_t)row * NFc + 3 * Dc;
  __shared__ float red[16];

  const int tid = threadIdx.x, lane = tid & 31, wave = tid >> 5;
  float sum = 0.f, sq = 0.f;
  for (int j = tid; j < FFc; j += 256) {
    float x = f[j];
    sum += x; sq += x * x;
  }
#pragma unroll
  for (int off = 16; off >= 1; off >>= 1) {
    sum += __shfl_xor(sum, off, 32);
    sq  += __shfl_xor(sq,  off, 32);
  }
  if (lane == 0) { red[wave] = sum; red[8 + wave] = sq; }
  __syncthreads();
  float ts = 0.f, tq = 0.f;
#pragma unroll
  for (int w = 0; w < 8; ++w) { ts += red[w]; tq += red[8 + w]; }
  const float mean = ts / FFc;
  const float rstd = rsqrtf(tq / FFc - mean * mean + EPSc);
  for (int j = tid; j < FFc; j += 256) {
    float x = (f[j] - mean) * rstd * gamma[j] + beta[j];
    Hb[(size_t)row * FFc + j] = (bf16)fmaxf(x, 0.f);
  }
}

// ---------------------------------------------------------------------------
// Causal flash attention. Grid: (S/128, B*H). 256 threads = 8 waves, each
// wave owns 16 q rows. K-tile = 32 keys staged in LDS; QK^T and P*V on WMMA.
// Output o as bf16 in [B,S,D] (heads re-interleaved) for the proj GEMM.
// ---------------------------------------------------------------------------
__global__ __launch_bounds__(256) void flash_attn_kernel(
    const bf16* __restrict__ Q, const bf16* __restrict__ K,
    const bf16* __restrict__ V, bf16* __restrict__ Obf) {
  __shared__ bf16 Ks[32][DHc + 16];       // (kv, d)
  __shared__ bf16 Vt[DHc][48];            // (d, kv)
  __shared__ bf16 Ps[8][16][48];          // per-wave P scratch (m, kv)

  const int tid  = threadIdx.x;
  const int lane = tid & 31;
  const int wave = tid >> 5;
  const int lrow = lane & 15;
  const int lkb  = (lane >> 4) * 8;
  const int lhi  = (lane >> 4) * 8;

  const int q0 = blockIdx.x * 128;
  const int b  = blockIdx.y / Hc;
  const int h  = blockIdx.y % Hc;
  const size_t bh = ((size_t)b * Hc + h) * Sc;

  v16bf qf[4];
  {
    const bf16* qbase = Q + (bh + q0 + wave * 16 + lrow) * DHc + lkb;
#pragma unroll
    for (int dc = 0; dc < 4; ++dc) qf[dc] = frag_ld(qbase + dc * 32);
  }

  float mrun[8], lrun[8];
  v8f   oacc[8];
#pragma unroll
  for (int g = 0; g < 8; ++g) { mrun[g] = -1e30f; lrun[g] = 0.f; }
#pragma unroll
  for (int n = 0; n < 8; ++n) oacc[n] = zero8();

  const int ntiles = q0 / 32 + 4;   // causal
  for (int kt = 0; kt < ntiles; ++kt) {
    const int kbase = kt * 32;
    {
      int r   = tid >> 4;
      int seg = (tid & 15) * 8;
#pragma unroll
      for (int half = 0; half < 2; ++half) {
        int kv = r + half * 16;
        *(uint4*)&Ks[kv][seg] = *(const uint4*)(K + (bh + kbase + kv) * DHc + seg);
        union { uint4 q; bf16 e[8]; } u;
        u.q = *(const uint4*)(V + (bh + kbase + kv) * DHc + seg);
#pragma unroll
        for (int e = 0; e < 8; ++e) Vt[seg + e][kv] = u.e[e];
      }
    }
    __syncthreads();

    v8f s[2] = {zero8(), zero8()};
#pragma unroll
    for (int kk = 0; kk < 2; ++kk)
#pragma unroll
      for (int dc = 0; dc < 4; ++dc) {
        v16bf kf = frag_ld(&Ks[kk * 16 + lrow][dc * 32 + lkb]);
        s[kk] = __builtin_amdgcn_wmma_f32_16x16x32_bf16(
            false, qf[dc], false, kf, (short)0, s[kk], false, false);
      }

#pragma unroll
    for (int kk = 0; kk < 2; ++kk) {
      int kidx = kbase + kk * 16 + lrow;
#pragma unroll
      for (int g = 0; g < 8; ++g) {
        int qidx = q0 + wave * 16 + g + lhi;
        if (kidx > qidx) s[kk][g] = -1e30f;
      }
    }

#pragma unroll
    for (int g = 0; g < 8; ++g) {
      float tmax = fmaxf(s[0][g], s[1][g]);
#pragma unroll
      for (int off = 8; off >= 1; off >>= 1)
        tmax = fmaxf(tmax, __shfl_xor(tmax, off, 32));
      float mnew  = fmaxf(mrun[g], tmax);
      float scale = __expf(mrun[g] - mnew);
      float p0 = __expf(s[0][g] - mnew);
      float p1 = __expf(s[1][g] - mnew);
      float rs = p0 + p1;
#pragma unroll
      for (int off = 8; off >= 1; off >>= 1) rs += __shfl_xor(rs, off, 32);
      lrun[g] = lrun[g] * scale + rs;
      mrun[g] = mnew;
      int prow = g + lhi;
      Ps[wave][prow][lrow]      = (bf16)p0;
      Ps[wave][prow][16 + lrow] = (bf16)p1;
#pragma unroll
      for (int n = 0; n < 8; ++n) oacc[n][g] *= scale;
    }

    v16bf pf = frag_ld(&Ps[wave][lrow][lkb]);
#pragma unroll
    for (int n = 0; n < 8; ++n) {
      v16bf vf = frag_ld(&Vt[n * 16 + lrow][lkb]);
      oacc[n] = __builtin_amdgcn_wmma_f32_16x16x32_bf16(
          false, pf, false, vf, (short)0, oacc[n], false, false);
    }
    __syncthreads();
  }

#pragma unroll
  for (int g = 0; g < 8; ++g) {
    int qrow = q0 + wave * 16 + g + lhi;
    float inv = 1.f / lrun[g];
    bf16* orow = Obf + ((size_t)b * Sc + qrow) * Dc + h * DHc;
#pragma unroll
    for (int n = 0; n < 8; ++n)
      orow[n * 16 + lrow] = (bf16)(oacc[n][g] * inv);
  }
}

// ---------------------------------------------------------------------------
// out = LayerNorm(attn + ffo*sigmoid(ffo); gamma2, beta2)   (f32 output)
// ---------------------------------------------------------------------------
__global__ __launch_bounds__(256) void final_ln_kernel(
    const float* __restrict__ attn, const float* __restrict__ ffo,
    const float* __restrict__ gamma, const float* __restrict__ beta,
    float* __restrict__ out) {
  const int row = blockIdx.x;
  const float* a = attn + (size_t)row * Dc;
  const float* f = ffo  + (size_t)row * Dc;
  float* o = out + (size_t)row * Dc;
  __shared__ float red[16];

  const int tid = threadIdx.x, lane = tid & 31, wave = tid >> 5;
  float y[8];
  float sum = 0.f, sq = 0.f;
#pragma unroll
  for (int i = 0; i < 8; ++i) {
    int j = tid + i * 256;
    float fv = f[j];
    float v  = a[j] + fv / (1.f + __expf(-fv));
    y[i] = v; sum += v; sq += v * v;
  }
#pragma unroll
  for (int off = 16; off >= 1; off >>= 1) {
    sum += __shfl_xor(sum, off, 32);
    sq  += __shfl_xor(sq,  off, 32);
  }
  if (lane == 0) { red[wave] = sum; red[8 + wave] = sq; }
  __syncthreads();
  float ts = 0.f, tq = 0.f;
#pragma unroll
  for (int w = 0; w < 8; ++w) { ts += red[w]; tq += red[8 + w]; }
  const float mean = ts / Dc;
  const float rstd = rsqrtf(tq / Dc - mean * mean + EPSc);
#pragma unroll
  for (int i = 0; i < 8; ++i) {
    int j = tid + i * 256;
    o[j] = (y[i] - mean) * rstd * gamma[j] + beta[j];
  }
}

// ---------------------------------------------------------------------------
// Host-side orchestration
// ---------------------------------------------------------------------------
extern "C" void kernel_launch(void* const* d_in, const int* in_sizes, int n_in,
                              void* d_out, int out_size, void* d_ws, size_t ws_size,
                              hipStream_t stream) {
  (void)in_sizes; (void)n_in; (void)out_size; (void)ws_size;
  const float* x       = (const float*)d_in[0];
  const float* W_fused = (const float*)d_in[1];
  const float* b_fused = (const float*)d_in[2];
  const float* W_attn  = (const float*)d_in[3];
  const float* b_attn  = (const float*)d_in[4];
  const float* W_ff    = (const float*)d_in[5];
  const float* b_ff    = (const float*)d_in[6];
  const float* gamma1  = (const float*)d_in[7];
  const float* beta1   = (const float*)d_in[8];
  const float* gamma2  = (const float*)d_in[9];
  const float* beta2   = (const float*)d_in[10];
  const float* rope_w  = (const float*)d_in[11];
  float* out = (float*)d_out;

  const size_t ROWS = (size_t)Bc * Sc;       // 4096
  char* ws = (char*)d_ws;
  size_t off = 0;
  auto carve = [&](size_t bytes) {
    char* p = ws + off;
    off += (bytes + 255) & ~(size_t)255;
    return p;
  };
  bf16*  xb    = (bf16*)carve(ROWS * Dc * sizeof(bf16));
  bf16*  wfubT = (bf16*)carve((size_t)NFc * Dc * sizeof(bf16));  // [NF, D]
  bf16*  wabT  = (bf16*)carve((size_t)Dc * Dc * sizeof(bf16));   // [D, D]
  bf16*  wffbT = (bf16*)carve((size_t)Dc * FFc * sizeof(bf16));  // [D, FF]
  float* fused = (float*)carve(ROWS * NFc * sizeof(float));
  bf16*  qb    = (bf16*)carve(ROWS * Dc * sizeof(bf16));
  bf16*  kb    = (bf16*)carve(ROWS * Dc * sizeof(bf16));
  bf16*  vb    = (bf16*)carve(ROWS * Dc * sizeof(bf16));
  bf16*  hb    = (bf16*)carve(ROWS * FFc * sizeof(bf16));
  bf16*  ob    = (bf16*)carve(ROWS * Dc * sizeof(bf16));
  float* attn  = (float*)carve(ROWS * Dc * sizeof(float));
  float* ffo   = (float*)carve(ROWS * Dc * sizeof(float));

  // activations: plain convert
  {
    size_t n = ROWS * Dc;
    cvt_bf16_kernel<<<4096, 256, 0, stream>>>(x, xb, n);
  }
  // weights: transpose + convert (GEMMs are NT)
  transpose_cvt_kernel<<<dim3(NFc / 32, Dc / 32), 256, 0, stream>>>(W_fused, wfubT, Dc, NFc);
  transpose_cvt_kernel<<<dim3(Dc / 32, Dc / 32), 256, 0, stream>>>(W_attn, wabT, Dc, Dc);
  transpose_cvt_kernel<<<dim3(Dc / 32, FFc / 32), 256, 0, stream>>>(W_ff, wffbT, FFc, Dc);

  // fused = x @ W_fused + b_fused         [4096 x 14336]
  gemm_bf16_nt_kernel<<<dim3(NFc / 128, ROWS / 128), 256, 0, stream>>>(
      xb, wfubT, b_fused, fused, (int)ROWS, NFc, Dc);

  // q,k,v (+RoPE) and LN+ReLU(ff) -> h
  qkv_rope_kernel<<<(int)ROWS, 256, 0, stream>>>(fused, rope_w, qb, kb, vb);
  ln_relu_kernel<<<(int)ROWS, 256, 0, stream>>>(fused, gamma1, beta1, hb);

  // causal flash attention -> o (bf16, [B,S,D])
  flash_attn_kernel<<<dim3(Sc / 128, Bc * Hc), 256, 0, stream>>>(qb, kb, vb, ob);

  // attn = o @ W_attn + b_attn            [4096 x 2048]
  gemm_bf16_nt_kernel<<<dim3(Dc / 128, ROWS / 128), 256, 0, stream>>>(
      ob, wabT, b_attn, attn, (int)ROWS, Dc, Dc);

  // ffo = h @ W_ff + b_ff                 [4096 x 2048]
  gemm_bf16_nt_kernel<<<dim3(Dc / 128, ROWS / 128), 256, 0, stream>>>(
      hb, wffbT, b_ff, ffo, (int)ROWS, Dc, FFc);

  // out = LN(attn + swish(ffo))
  final_ln_kernel<<<(int)ROWS, 256, 0, stream>>>(attn, ffo, gamma2, beta2, out);
}
